// LayoutTransformerEncoder_83365315215546
// MI455X (gfx1250) — matrix operand order repaired and checked by reading, not compiled, not run on hardware
//
#include <hip/hip_runtime.h>
#include <math.h>

// ---------------------------------------------------------------------------
// Types for CDNA5 WMMA (wave32): 16x16x32 bf16 -> f32
// ---------------------------------------------------------------------------
typedef __attribute__((ext_vector_type(16))) __bf16 bfrag;
typedef __attribute__((ext_vector_type(8)))  float  f8;

union FragU {
    unsigned int u[8];
    uint4        q[2];
    bfrag        v;
};

__device__ __forceinline__ f8 f8zero() {
    f8 z;
#pragma unroll
    for (int i = 0; i < 8; ++i) z[i] = 0.0f;
    return z;
}

__device__ __forceinline__ unsigned short f2bf(float f) {
    unsigned int u = __float_as_uint(f);
    u += 0x7FFFu + ((u >> 16) & 1u);   // round-to-nearest-even
    return (unsigned short)(u >> 16);
}

__device__ __forceinline__ f8 wmma_bf16(bfrag a, bfrag b, f8 c) {
    return __builtin_amdgcn_wmma_f32_16x16x32_bf16(
        false, a, false, b, (short)0, c, false, false);
}

// A-matrix 16x32 bf16 fragment from LDS (row-major tile).
// rowp = &tile[row * stride + kTileOffset]; kb = (lane>>4)*8.
// Contiguous K order (matches B): lane<16 K=0..15; lane>=16 K=16..31.
__device__ __forceinline__ bfrag a_frag_from(const unsigned short* rowp, int kb) {
    FragU f;
#pragma unroll
    for (int pu = 0; pu < 4; ++pu)
        f.u[pu] = *(const unsigned int*)(rowp + 2 * kb + 2 * pu);
#pragma unroll
    for (int pu = 0; pu < 4; ++pu)
        f.u[4 + pu] = *(const unsigned int*)(rowp + 2 * kb + 8 + 2 * pu);
    return f.v;
}

// B-matrix fragment when the 16 K-values are contiguous in LDS.
// Contiguous K order (matches A): lanes 0-15 carry K = k0..k0+15, lanes 16-31 K = k0+16..31;
// caller passes p = &tileT[n * stride + k0 + (lane>>4)*16].
__device__ __forceinline__ bfrag b_frag_contig(const unsigned short* p) {
    FragU f;
    f.q[0] = *(const uint4*)(p);
    f.q[1] = *(const uint4*)(p + 8);
    return f.v;
}

// ---------------------------------------------------------------------------
// Embedding fusion: pos + class_table[class] + bbox@W + bbox2d@W2 + biases
// ---------------------------------------------------------------------------
__global__ __launch_bounds__(256) void embed_kernel(
    const float* __restrict__ obj_bbox, const float* __restrict__ obj_bbox2d,
    const int* __restrict__ obj_class, const float* __restrict__ pos_emb,
    const float* __restrict__ class_table, const float* __restrict__ W_bbox,
    const float* __restrict__ b_bbox, const float* __restrict__ W_bbox2d,
    const float* __restrict__ b_bbox2d, float* __restrict__ x)
{
    const int row = blockIdx.x;          // b*T + t
    const int t   = row & 255;
    const int cls = obj_class[row];
    float bb[8], b2[4];
#pragma unroll
    for (int i = 0; i < 8; ++i) bb[i] = obj_bbox[row * 8 + i];
#pragma unroll
    for (int i = 0; i < 4; ++i) b2[i] = obj_bbox2d[row * 4 + i];
#pragma unroll
    for (int i = 0; i < 4; ++i) {
        const int dd = threadIdx.x + i * 256;
        float v = pos_emb[t * 1024 + dd] + class_table[cls * 1024 + dd]
                + b_bbox[dd] + b_bbox2d[dd];
#pragma unroll
        for (int k = 0; k < 8; ++k) v += bb[k] * W_bbox[k * 1024 + dd];
#pragma unroll
        for (int k = 0; k < 4; ++k) v += b2[k] * W_bbox2d[k * 1024 + dd];
        x[(size_t)row * 1024 + dd] = v;
    }
}

// ---------------------------------------------------------------------------
// LayerNorm over D=1024; MODE 0 -> bf16 out, MODE 1 -> f32 out
// ---------------------------------------------------------------------------
template <int MODE>
__global__ __launch_bounds__(256) void ln_kernel(
    const float* __restrict__ x, const float* __restrict__ g,
    const float* __restrict__ bta, unsigned short* __restrict__ outb,
    float* __restrict__ outf)
{
    const int row = blockIdx.x;
    const float* xr = x + (size_t)row * 1024;
    float vals[4];
    float s = 0.f, ss = 0.f;
#pragma unroll
    for (int i = 0; i < 4; ++i) {
        float v = xr[threadIdx.x + i * 256];
        vals[i] = v; s += v; ss += v * v;
    }
#pragma unroll
    for (int m = 1; m < 32; m <<= 1) {
        s  += __shfl_xor(s,  m, 32);
        ss += __shfl_xor(ss, m, 32);
    }
    __shared__ float sbuf[16];
    const int w = threadIdx.x >> 5;
    if ((threadIdx.x & 31) == 0) { sbuf[w] = s; sbuf[8 + w] = ss; }
    __syncthreads();
    s = 0.f; ss = 0.f;
#pragma unroll
    for (int i = 0; i < 8; ++i) { s += sbuf[i]; ss += sbuf[8 + i]; }
    const float mean = s * (1.0f / 1024.0f);
    const float var  = ss * (1.0f / 1024.0f) - mean * mean;
    const float rinv = rsqrtf(var + 1e-5f);
#pragma unroll
    for (int i = 0; i < 4; ++i) {
        const int dd = threadIdx.x + i * 256;
        const float y = (vals[i] - mean) * rinv * g[dd] + bta[dd];
        if (MODE == 0) outb[(size_t)row * 1024 + dd] = f2bf(y);
        else           outf[(size_t)row * 1024 + dd] = y;
    }
}

// ---------------------------------------------------------------------------
// WMMA GEMM: C[M,N] = A_bf16[M,K] * B_f32->bf16[K,N] (+ bias, epilogues)
//   EPI 0: out_bf16 = acc + bias
//   EPI 1: out_f32 += acc + bias        (residual accumulate into x)
//   EPI 2: out_bf16 = gelu(acc + bias)
// Block tile 128x64, 8 waves (4x2), wave tile 32x32, K-step 32.
// B tile stored TRANSPOSED in LDS ([N][K]) so B fragments are 2x ds_load_b128.
// ---------------------------------------------------------------------------
#define STRA  40   // A tile row stride (32 + 8 pad), conflict-free frag reads
#define STRBT 40   // B^T tile row stride (32 + 8 pad)

template <int EPI>
__global__ __launch_bounds__(256) void gemm_kernel(
    const unsigned short* __restrict__ A, const float* __restrict__ Bw,
    const float* __restrict__ bias, unsigned short* __restrict__ outB,
    float* __restrict__ outF, int M, int N, int K)
{
    __shared__ unsigned short As[128 * STRA];
    __shared__ unsigned short BsT[64 * STRBT];
    const int tid  = threadIdx.x;
    const int lane = tid & 31, w = tid >> 5;
    const int wm = w >> 1, wn = w & 1;
    const int m0 = blockIdx.y * 128, n0 = blockIdx.x * 64;
    const int lm = lane & 15, lh = lane >> 4;

    f8 acc[2][2];
#pragma unroll
    for (int i = 0; i < 2; ++i)
#pragma unroll
        for (int j = 0; j < 2; ++j) acc[i][j] = f8zero();

    const int arow = tid >> 1, ahalf = tid & 1;
    const int brow = tid >> 3, bseg  = tid & 7;

    for (int k0 = 0; k0 < K; k0 += 32) {
        {   // A tile 128x32 bf16, coalesced 16B chunks
            const uint4* src = (const uint4*)(A + (size_t)(m0 + arow) * K + k0 + ahalf * 16);
            uint4* dst = (uint4*)(As + arow * STRA + ahalf * 16);
            dst[0] = src[0];
            dst[1] = src[1];
        }
        {   // B tile 32x64: fp32 -> bf16, stored transposed [n][k]
            const float4* src = (const float4*)(Bw + (size_t)(k0 + brow) * N + n0 + bseg * 8);
            float4 f0 = src[0], f1 = src[1];
            unsigned short cv[8];
            cv[0] = f2bf(f0.x); cv[1] = f2bf(f0.y); cv[2] = f2bf(f0.z); cv[3] = f2bf(f0.w);
            cv[4] = f2bf(f1.x); cv[5] = f2bf(f1.y); cv[6] = f2bf(f1.z); cv[7] = f2bf(f1.w);
#pragma unroll
            for (int i = 0; i < 8; ++i)
                BsT[(bseg * 8 + i) * STRBT + brow] = cv[i];
        }
        // Prefetch next K-slice into cache while this slice is consumed.
        if (k0 + 32 < K) {
            __builtin_prefetch(A + (size_t)(m0 + arow) * K + (k0 + 32) + ahalf * 16, 0, 3);
            __builtin_prefetch(Bw + (size_t)(k0 + 32 + brow) * N + n0 + bseg * 8, 0, 3);
        }
        __syncthreads();

        bfrag afr[2], bfr[2];
#pragma unroll
        for (int i = 0; i < 2; ++i)
            afr[i] = a_frag_from(As + (wm * 32 + i * 16 + lm) * STRA, lh * 8);
#pragma unroll
        for (int j = 0; j < 2; ++j)
            bfr[j] = b_frag_contig(BsT + (wn * 32 + j * 16 + lm) * STRBT + lh * 16);
#pragma unroll
        for (int i = 0; i < 2; ++i)
#pragma unroll
            for (int j = 0; j < 2; ++j)
                acc[i][j] = wmma_bf16(afr[i], bfr[j], acc[i][j]);
        __syncthreads();
    }

#pragma unroll
    for (int i = 0; i < 2; ++i) {
#pragma unroll
        for (int j = 0; j < 2; ++j) {
            const int col = n0 + wn * 32 + j * 16 + lm;
            const float bv = bias[col];
#pragma unroll
            for (int v = 0; v < 8; ++v) {
                const int row = m0 + wm * 32 + i * 16 + lh * 8 + v;
                const size_t idx = (size_t)row * N + col;
                const float val = acc[i][j][v] + bv;
                if (EPI == 0) {
                    outB[idx] = f2bf(val);
                } else if (EPI == 1) {
                    outF[idx] += val;
                } else {
                    const float gv = 0.5f * val * (1.0f + erff(val * 0.70710678118f));
                    outB[idx] = f2bf(gv);
                }
            }
        }
    }
}

// ---------------------------------------------------------------------------
// Attention: one block per (head, batch). T=256, C=64, softmax scale 1/8.
// qkv bf16 [B*T, 3072]; out bf16 [B*T, 1024] at columns hh*64..hh*64+63.
// q,k staged row-major; v staged TRANSPOSED ([n][k]) for contiguous B frags.
// ---------------------------------------------------------------------------
#define QSTR 72    // q/k row stride (64 + 8 pad)
#define VSTR 264   // v^T / P row stride (256 + 8 pad)
#define PSTR 264

__global__ __launch_bounds__(256) void attn_kernel(
    const unsigned short* __restrict__ qkv, unsigned short* __restrict__ aout)
{
    extern __shared__ unsigned short sm[];
    unsigned short* qs  = sm;                            // [256][QSTR]
    unsigned short* ks  = sm + 256 * QSTR;               // [256][QSTR]
    unsigned short* vsT = sm + 2 * 256 * QSTR;           // [64][VSTR]
    unsigned short* ps  = sm + 2 * 256 * QSTR + 64 * VSTR; // per-wave 16x256 strips

    const int hh = blockIdx.x, bb = blockIdx.y;
    const int tid = threadIdx.x, lane = tid & 31, w = tid >> 5;
    const int lm = lane & 15, lh = lane >> 4;
    const size_t rowbase = (size_t)(bb * 256) * 3072;

    // ---- stage q, k row-major ----
#pragma unroll
    for (int sel = 0; sel < 2; ++sel) {
        unsigned short* dst = (sel == 0) ? qs : ks;
        const int colOff = hh * 192 + sel * 64;
        for (int c = tid; c < 2048; c += 256) {
            const int r = c >> 3, seg = c & 7;
            const uint4* src = (const uint4*)(qkv + rowbase + (size_t)r * 3072 + colOff + seg * 8);
            *(uint4*)(dst + r * QSTR + seg * 8) = *src;
        }
    }
    // ---- stage v transposed: vsT[n][k] ----
    {
        const int colOff = hh * 192 + 128;
        for (int c = tid; c < 2048; c += 256) {
            const int r = c >> 3, seg = c & 7;
            union { uint4 q; unsigned short s[8]; } u;
            u.q = *(const uint4*)(qkv + rowbase + (size_t)r * 3072 + colOff + seg * 8);
#pragma unroll
            for (int i = 0; i < 8; ++i)
                vsT[(seg * 8 + i) * VSTR + r] = u.s[i];
        }
    }
    __syncthreads();

    for (int pass = 0; pass < 2; ++pass) {
        const int r0 = pass * 128 + w * 16;

        // ---- S = q @ k^T over full row strip (16 x 256), K=64 ----
        const bfrag qa0 = a_frag_from(qs + (r0 + lm) * QSTR + 0,  lh * 8);
        const bfrag qa1 = a_frag_from(qs + (r0 + lm) * QSTR + 32, lh * 8);
        f8 sacc[16];
#pragma unroll
        for (int j = 0; j < 16; ++j) sacc[j] = f8zero();
#pragma unroll
        for (int j = 0; j < 16; ++j) {
            const bfrag kb0 = b_frag_contig(ks + (j * 16 + lm) * QSTR + 0  + lh * 16);
            const bfrag kb1 = b_frag_contig(ks + (j * 16 + lm) * QSTR + 32 + lh * 16);
            sacc[j] = wmma_bf16(qa0, kb0, sacc[j]);
            sacc[j] = wmma_bf16(qa1, kb1, sacc[j]);
        }

        // ---- softmax over 256 cols (rows live in 16-lane halves) ----
        float rmax[8], rsum[8];
#pragma unroll
        for (int v = 0; v < 8; ++v) {
            float m = -1e30f;
#pragma unroll
            for (int j = 0; j < 16; ++j) m = fmaxf(m, sacc[j][v]);
#pragma unroll
            for (int msk = 1; msk < 16; msk <<= 1) m = fmaxf(m, __shfl_xor(m, msk, 32));
            rmax[v] = m;
            rsum[v] = 0.0f;
        }
#pragma unroll
        for (int j = 0; j < 16; ++j) {
#pragma unroll
            for (int v = 0; v < 8; ++v) {
                const float p = __expf(0.125f * (sacc[j][v] - rmax[v]));
                sacc[j][v] = p;
                rsum[v] += p;
            }
        }
#pragma unroll
        for (int v = 0; v < 8; ++v) {
            float s = rsum[v];
#pragma unroll
            for (int msk = 1; msk < 16; msk <<= 1) s += __shfl_xor(s, msk, 32);
            rsum[v] = 1.0f / s;
        }

        // ---- write bf16 probabilities to per-wave LDS strip ----
        unsigned short* psw = ps + w * 16 * PSTR;
#pragma unroll
        for (int j = 0; j < 16; ++j) {
#pragma unroll
            for (int v = 0; v < 8; ++v) {
                const int lr = lh * 8 + v;
                psw[lr * PSTR + j * 16 + lm] = f2bf(sacc[j][v] * rsum[v]);
            }
        }
        __syncthreads();

        // ---- O = P @ V : 16 x 64, K=256 in steps of 32 ----
        f8 oacc[4];
#pragma unroll
        for (int n2 = 0; n2 < 4; ++n2) oacc[n2] = f8zero();
#pragma unroll
        for (int kk = 0; kk < 8; ++kk) {
            const bfrag pa = a_frag_from(psw + lm * PSTR + kk * 32, lh * 8);
#pragma unroll
            for (int n2 = 0; n2 < 4; ++n2) {
                const bfrag vb =
                    b_frag_contig(vsT + (n2 * 16 + lm) * VSTR + kk * 32 + lh * 16);
                oacc[n2] = wmma_bf16(pa, vb, oacc[n2]);
            }
        }
#pragma unroll
        for (int n2 = 0; n2 < 4; ++n2) {
            const int col = hh * 64 + n2 * 16 + lm;
#pragma unroll
            for (int v = 0; v < 8; ++v) {
                const int row = bb * 256 + r0 + lh * 8 + v;
                aout[(size_t)row * 1024 + col] = f2bf(oacc[n2][v]);
            }
        }
        __syncthreads();
    }
}

// ---------------------------------------------------------------------------
// xf_proj = y[:, token 0] @ W_proj + b_proj   (64 x 1024, tiny)
// ---------------------------------------------------------------------------
__global__ __launch_bounds__(256) void proj_kernel(
    const float* __restrict__ y, const float* __restrict__ Wp,
    const float* __restrict__ bp, float* __restrict__ out)
{
    const int b = blockIdx.x;
    __shared__ float yr[1024];
    for (int i = threadIdx.x; i < 1024; i += 256)
        yr[i] = y[((size_t)b * 256) * 1024 + i];
    __syncthreads();
#pragma unroll
    for (int j = 0; j < 4; ++j) {
        const int o = threadIdx.x + j * 256;
        float acc = bp[o];
        for (int d = 0; d < 1024; ++d) acc += yr[d] * Wp[(size_t)d * 1024 + o];
        out[b * 1024 + o] = acc;
    }
}

// ---------------------------------------------------------------------------
// [B,T,D] -> [B,D,T] tiled transpose
// ---------------------------------------------------------------------------
__global__ void transpose_kernel(const float* __restrict__ y, float* __restrict__ out)
{
    __shared__ float tile[32][33];
    const int b  = blockIdx.z;
    const int d0 = blockIdx.x * 32, t0 = blockIdx.y * 32;
#pragma unroll
    for (int i = 0; i < 4; ++i) {
        const int tt = t0 + threadIdx.y + i * 8;
        tile[threadIdx.y + i * 8][threadIdx.x] =
            y[((size_t)(b * 256 + tt)) * 1024 + d0 + threadIdx.x];
    }
    __syncthreads();
#pragma unroll
    for (int i = 0; i < 4; ++i) {
        const int dd = d0 + threadIdx.y + i * 8;
        out[((size_t)(b * 1024 + dd)) * 256 + t0 + threadIdx.x] =
            tile[threadIdx.x][threadIdx.y + i * 8];
    }
}

// ---------------------------------------------------------------------------
// Host orchestration
// ---------------------------------------------------------------------------
extern "C" void kernel_launch(void* const* d_in, const int* in_sizes, int n_in,
                              void* d_out, int out_size, void* d_ws, size_t ws_size,
                              hipStream_t stream)
{
    (void)in_sizes; (void)n_in; (void)out_size; (void)ws_size;

    const float* obj_bbox    = (const float*)d_in[0];
    const float* obj_bbox2d  = (const float*)d_in[1];
    const int*   obj_class   = (const int*)d_in[2];
    const float* pos_emb     = (const float*)d_in[3];
    const float* class_table = (const float*)d_in[4];
    const float* W_bbox      = (const float*)d_in[5];
    const float* b_bbox      = (const float*)d_in[6];
    const float* W_bbox2d    = (const float*)d_in[7];
    const float* b_bbox2d    = (const float*)d_in[8];
    const float* W_qkv = (const float*)d_in[9];
    const float* b_qkv = (const float*)d_in[10];
    const float* W_o   = (const float*)d_in[11];
    const float* b_o   = (const float*)d_in[12];
    const float* W_fc  = (const float*)d_in[13];
    const float* b_fc  = (const float*)d_in[14];
    const float* W_pj  = (const float*)d_in[15];
    const float* b_pj  = (const float*)d_in[16];
    const float* ln1_g = (const float*)d_in[17];
    const float* ln1_b = (const float*)d_in[18];
    const float* ln2_g = (const float*)d_in[19];
    const float* ln2_b = (const float*)d_in[20];
    const float* fin_g = (const float*)d_in[21];
    const float* fin_b = (const float*)d_in[22];
    const float* W_proj = (const float*)d_in[23];
    const float* b_proj = (const float*)d_in[24];

    char* ws = (char*)d_ws;
    float*          x    = (float*)(ws + 0);                    // 64 MB  f32 [16384,1024]
    float*          yfin = (float*)(ws + 67108864);             // 64 MB  f32
    unsigned short* qkvb = (unsigned short*)(ws + 134217728);   // 96 MB  bf16 [16384,3072]
    unsigned short* hb   = (unsigned short*)(ws + 234881024);   // 32 MB  bf16
    unsigned short* ab   = (unsigned short*)(ws + 268435456);   // 32 MB  bf16
    unsigned short* ub   = (unsigned short*)(ws + 301989888);   // 128 MB bf16 [16384,4096]

    float* outp = (float*)d_out;
    float* outx = outp + 64 * 1024;

    embed_kernel<<<16384, 256, 0, stream>>>(obj_bbox, obj_bbox2d, obj_class, pos_emb,
        class_table, W_bbox, b_bbox, W_bbox2d, b_bbox2d, x);

    const size_t attn_lds =
        (size_t)(2 * 256 * QSTR + 64 * VSTR + 8 * 16 * PSTR) * 2;

    for (int l = 0; l < 8; ++l) {
        const float* Wqkv_l = W_qkv + (size_t)l * 1024 * 3072;
        const float* bqkv_l = b_qkv + (size_t)l * 3072;
        const float* Wo_l   = W_o   + (size_t)l * 1024 * 1024;
        const float* bo_l   = b_o   + (size_t)l * 1024;
        const float* Wfc_l  = W_fc  + (size_t)l * 1024 * 4096;
        const float* bfc_l  = b_fc  + (size_t)l * 4096;
        const float* Wpj_l  = W_pj  + (size_t)l * 4096 * 1024;
        const float* bpj_l  = b_pj  + (size_t)l * 1024;

        ln_kernel<0><<<16384, 256, 0, stream>>>(x, ln1_g + l * 1024, ln1_b + l * 1024, hb, nullptr);
        gemm_kernel<0><<<dim3(3072 / 64, 16384 / 128), 256, 0, stream>>>(
            hb, Wqkv_l, bqkv_l, qkvb, nullptr, 16384, 3072, 1024);
        attn_kernel<<<dim3(16, 64), 256, attn_lds, stream>>>(qkvb, ab);
        gemm_kernel<1><<<dim3(1024 / 64, 16384 / 128), 256, 0, stream>>>(
            ab, Wo_l, bo_l, nullptr, x, 16384, 1024, 1024);
        ln_kernel<0><<<16384, 256, 0, stream>>>(x, ln2_g + l * 1024, ln2_b + l * 1024, hb, nullptr);
        gemm_kernel<2><<<dim3(4096 / 64, 16384 / 128), 256, 0, stream>>>(
            hb, Wfc_l, bfc_l, ub, nullptr, 16384, 4096, 1024);
        gemm_kernel<1><<<dim3(1024 / 64, 16384 / 128), 256, 0, stream>>>(
            ub, Wpj_l, bpj_l, nullptr, x, 16384, 1024, 4096);
    }

    ln_kernel<1><<<16384, 256, 0, stream>>>(x, fin_g, fin_b, nullptr, yfin);
    proj_kernel<<<64, 256, 0, stream>>>(yfin, W_proj, b_proj, outp);
    transpose_kernel<<<dim3(32, 8, 64), dim3(32, 8), 0, stream>>>(yfin, outx);
}
